// Decoder_TacotronOne_42090679501592
// MI455X (gfx1250) — matrix operand order repaired and checked by reading, not compiled, not run on hardware
//
#include <hip/hip_runtime.h>
#include <stdint.h>

typedef unsigned short u16;
typedef __attribute__((ext_vector_type(16))) __bf16 v16bf;
typedef __attribute__((ext_vector_type(8)))  float  v8f;

#define LOG2E 1.44269504088896f

__device__ __forceinline__ u16 f2bf(float f){
  unsigned u = __builtin_bit_cast(unsigned, f);
  unsigned r = u + 0x7FFFu + ((u >> 16) & 1u);   // RNE
  return (u16)(r >> 16);
}
__device__ __forceinline__ float bf2f(u16 h){
  unsigned u = ((unsigned)h) << 16;
  return __builtin_bit_cast(float, u);
}
__device__ __forceinline__ float fast_exp(float x){ return __builtin_amdgcn_exp2f(x * LOG2E); }
__device__ __forceinline__ float fast_sig(float x){
  float e = __builtin_amdgcn_exp2f(-x * LOG2E);
  return __builtin_amdgcn_rcpf(1.0f + e);
}
__device__ __forceinline__ float fast_tanh(float x){
  x = fminf(15.0f, fmaxf(-15.0f, x));
  float e = __builtin_amdgcn_exp2f(x * (2.0f * LOG2E));
  return (e - 1.0f) * __builtin_amdgcn_rcpf(e + 1.0f);
}

union FragU { uint4 q[2]; v16bf v; };

// A fragment 16x32 bf16: lane<16 -> row M=lane, K {0..7}+{16..23}; lane>=16 -> K {8..15}+{24..31}
__device__ __forceinline__ v16bf load_fragA(const u16* A, int lda, int mBase, int kofs, int lane){
  const u16* p = A + (size_t)(mBase + (lane & 15)) * lda + kofs + ((lane & 16) ? 8 : 0);
  FragU f;
  f.q[0] = *(const uint4*)(p);
  f.q[1] = *(const uint4*)(p + 16);
  return f.v;
}
// B fragment 32x16 (B[k,n] = W[n,k]): lane<16 -> col N=lane, K {0..7}+{16..23}; lane>=16 -> K {8..15}+{24..31}
__device__ __forceinline__ v16bf load_fragB(const u16* W, int ldw, int nBase, int kofs, int lane){
  const u16* p = W + (size_t)(nBase + (lane & 15)) * ldw + kofs + ((lane & 16) ? 8 : 0);
  FragU f;
  f.q[0] = *(const uint4*)(p);
  f.q[1] = *(const uint4*)(p + 16);
  return f.v;
}

// One 16x16 tile of C = [A0|A1] @ W.T (+bias)(+relu). K0/K1 = k-steps from A0/A1 (compile-time).
template<int K0, int K1>
__device__ __forceinline__ void gemm_tile(
    const u16* __restrict__ A0, int lda0, const u16* __restrict__ A1, int lda1,
    const u16* __restrict__ W, int ldw, const float* __restrict__ bias,
    float* __restrict__ C, int ldc, u16* __restrict__ Cbf, int ldcbf,
    int mt, int nt, bool relu, int lane)
{
  v8f acc = {0.f,0.f,0.f,0.f,0.f,0.f,0.f,0.f};
  int mBase = mt * 16, nBase = nt * 16;
#pragma unroll
  for (int kt = 0; kt < K0; ++kt){
    v16bf a = load_fragA(A0, lda0, mBase, kt * 32, lane);
    v16bf b = load_fragB(W, ldw, nBase, kt * 32, lane);
    acc = __builtin_amdgcn_wmma_f32_16x16x32_bf16(false, a, false, b, (short)0, acc, false, false);
  }
#pragma unroll
  for (int kt = 0; kt < K1; ++kt){
    v16bf a = load_fragA(A1, lda1, mBase, kt * 32, lane);
    v16bf b = load_fragB(W, ldw, nBase, (K0 + kt) * 32, lane);
    acc = __builtin_amdgcn_wmma_f32_16x16x32_bf16(false, a, false, b, (short)0, acc, false, false);
  }
  int col = nBase + (lane & 15);
  float bv = bias ? bias[col] : 0.0f;
  int rBase = mBase + ((lane & 16) ? 8 : 0);
#pragma unroll
  for (int r = 0; r < 8; ++r){
    float val = acc[r] + bv;
    if (relu) val = fmaxf(val, 0.0f);
    size_t row = (size_t)(rBase + r);
    if (C)   C[row * (size_t)ldc + col] = val;
    if (Cbf) Cbf[row * (size_t)ldcbf + col] = f2bf(val);
  }
}

template<int K0, int K1>
__global__ void wmma_gemm_kernel(
    const u16* __restrict__ A0, int lda0, const u16* __restrict__ A1, int lda1,
    const u16* __restrict__ W, int ldw, const float* __restrict__ bias,
    float* __restrict__ C, int ldc, u16* __restrict__ Cbf, int ldcbf,
    int Mt, int Nt, int relu)
{
  int wave = threadIdx.x >> 5, lane = threadIdx.x & 31;
  int tile = blockIdx.x * (blockDim.x >> 5) + wave;
  if (tile >= Mt * Nt) return;
  gemm_tile<K0, K1>(A0, lda0, A1, lda1, W, ldw, bias, C, ldc, Cbf, ldcbf,
                    tile / Nt, tile % Nt, relu != 0, lane);
}

// GI = [Ai0|Ai1] @ Wi.T + bi ; GH = H @ Wh.T + bh   (both Mt*16 x Nt*16), one launch
template<int KI0, int KI1, int KH>
__global__ void wmma_gates_kernel(
    const u16* __restrict__ Ai0, int ldai0, const u16* __restrict__ Ai1, int ldai1,
    const u16* __restrict__ Wi, const float* __restrict__ bi,
    const u16* __restrict__ H, const u16* __restrict__ Wh, const float* __restrict__ bh,
    float* __restrict__ GI, float* __restrict__ GH, int Mt, int Nt)
{
  int wave = threadIdx.x >> 5, lane = threadIdx.x & 31;
  int tile = blockIdx.x * (blockDim.x >> 5) + wave;
  int half = Mt * Nt;
  if (tile >= 2 * half) return;
  if (tile < half){
    gemm_tile<KI0, KI1>(Ai0, ldai0, Ai1, ldai1, Wi, (KI0 + KI1) * 32, bi,
                        GI, Nt * 16, nullptr, 0, tile / Nt, tile % Nt, false, lane);
  } else {
    tile -= half;
    gemm_tile<KH, 0>(H, KH * 32, nullptr, 0, Wh, KH * 32, bh,
                     GH, Nt * 16, nullptr, 0, tile / Nt, tile % Nt, false, lane);
  }
}

// GRU elementwise update; optional residual dec += h'
__global__ void gru_update_kernel(const float* __restrict__ gi, const float* __restrict__ gh,
                                  float* __restrict__ h, u16* __restrict__ h_bf,
                                  float* __restrict__ dec, u16* __restrict__ dec_bf)
{
  int b = blockIdx.x, d = threadIdx.x;
  size_t g = (size_t)b * 768;
  float ir = gi[g + d], iz = gi[g + 256 + d], in_ = gi[g + 512 + d];
  float hr = gh[g + d], hz = gh[g + 256 + d], hn  = gh[g + 512 + d];
  float r = fast_sig(ir + hr);
  float z = fast_sig(iz + hz);
  float n = fast_tanh(in_ + r * hn);
  int i = b * 256 + d;
  float hp = (1.0f - z) * n + z * h[i];
  h[i] = hp; h_bf[i] = f2bf(hp);
  if (dec){ float dv = dec[i] + hp; dec[i] = dv; dec_bf[i] = f2bf(dv); }
}

// Per-batch: scores = v . tanh(q + pm), mask, softmax, align out, context attn
__global__ void attention_kernel(
    const float* __restrict__ q, const u16* __restrict__ pm, const u16* __restrict__ enc,
    const float* __restrict__ v, const int* __restrict__ mlen,
    float* __restrict__ alignOut, float* __restrict__ attn_f32, u16* __restrict__ attn_bf)
{
  __shared__ float qv[256];
  __shared__ float vv[256];
  __shared__ float sc[1024];
  __shared__ float red[256];
  int b = blockIdx.x, tid = threadIdx.x;
  qv[tid] = q[b * 256 + tid];
  vv[tid] = v[tid];
  __syncthreads();
  int len = mlen[b];
  float lmax = -3.0e38f;
  for (int tt = tid; tt < 1024; tt += 256){
    float s = -3.0e38f;
    if (tt < len){
      const u16* row = pm + ((size_t)b * 1024 + tt) * 256;
      float acc = 0.f;
      for (int d = 0; d < 256; d += 2){
        unsigned u = *(const unsigned*)(row + d);
        acc += vv[d]     * fast_tanh(qv[d]     + bf2f((u16)(u & 0xffffu)));
        acc += vv[d + 1] * fast_tanh(qv[d + 1] + bf2f((u16)(u >> 16)));
      }
      s = acc;
    }
    sc[tt] = s;
    lmax = fmaxf(lmax, s);
  }
  red[tid] = lmax;
  __syncthreads();
  for (int off = 128; off > 0; off >>= 1){
    if (tid < off) red[tid] = fmaxf(red[tid], red[tid + off]);
    __syncthreads();
  }
  float m = red[0];
  __syncthreads();
  float lsum = 0.f;
  for (int tt = tid; tt < 1024; tt += 256){
    float e = (tt < len) ? fast_exp(sc[tt] - m) : 0.f;
    sc[tt] = e;
    lsum += e;
  }
  red[tid] = lsum;
  __syncthreads();
  for (int off = 128; off > 0; off >>= 1){
    if (tid < off) red[tid] += red[tid + off];
    __syncthreads();
  }
  float inv = __builtin_amdgcn_rcpf(red[0]);
  __syncthreads();
  for (int tt = tid; tt < 1024; tt += 256){
    float a = sc[tt] * inv;
    sc[tt] = a;
    alignOut[(size_t)b * 409600 + tt] = a;   // 400*1024 per batch
  }
  __syncthreads();
  float acc = 0.f;
  const u16* eb = enc + (size_t)b * 1024 * 256 + tid;
#pragma unroll 4
  for (int tt = 0; tt < 1024; ++tt)
    acc += sc[tt] * bf2f(eb[(size_t)tt * 256]);
  attn_f32[b * 256 + tid] = acc;
  attn_bf[b * 256 + tid]  = f2bf(acc);
}

__global__ void convert_pad_kernel(const float* __restrict__ src, u16* __restrict__ dst,
                                   int rows, int K, int Kpad){
  size_t total = (size_t)rows * Kpad;
  for (size_t i = (size_t)blockIdx.x * blockDim.x + threadIdx.x; i < total;
       i += (size_t)gridDim.x * blockDim.x){
    int r = (int)(i / Kpad), k = (int)(i % Kpad);
    dst[i] = f2bf(k < K ? src[(size_t)r * K + k] : 0.f);
  }
}

// dec_in[t*64+b][k] = (t==0 || k>=400) ? 0 : inputs[b, t-1, k]  (Kpad=416)
__global__ void decin_kernel(const float* __restrict__ inp, u16* __restrict__ dst){
  size_t total = (size_t)25600 * 416;
  for (size_t i = (size_t)blockIdx.x * blockDim.x + threadIdx.x; i < total;
       i += (size_t)gridDim.x * blockDim.x){
    int g = (int)(i / 416), k = (int)(i % 416);
    int t = g >> 6, b = g & 63;
    float v = 0.f;
    if (t > 0 && k < 400) v = inp[((size_t)b * 400 + (t - 1)) * 400 + k];
    dst[i] = f2bf(v);
  }
}

__global__ void zero_kernel(unsigned* __restrict__ ptr, int n){
  int i = blockIdx.x * blockDim.x + threadIdx.x;
  if (i < n) ptr[i] = 0u;
}

extern "C" void kernel_launch(void* const* d_in, const int* in_sizes, int n_in,
                              void* d_out, int out_size, void* d_ws, size_t ws_size,
                              hipStream_t stream)
{
  (void)in_sizes; (void)n_in; (void)out_size; (void)ws_size;
  const float* enc    = (const float*)d_in[0];
  const float* inputs = (const float*)d_in[1];
  const int*   mlen   = (const int*)  d_in[2];
  const float* pre_w1 = (const float*)d_in[3];  const float* pre_b1 = (const float*)d_in[4];
  const float* pre_w2 = (const float*)d_in[5];  const float* pre_b2 = (const float*)d_in[6];
  const float* agru_wi= (const float*)d_in[7];  const float* agru_wh= (const float*)d_in[8];
  const float* agru_bi= (const float*)d_in[9];  const float* agru_bh= (const float*)d_in[10];
  const float* attn_qw= (const float*)d_in[11]; const float* attn_v = (const float*)d_in[12];
  const float* mem_w  = (const float*)d_in[13];
  const float* proj_w = (const float*)d_in[14]; const float* proj_b = (const float*)d_in[15];
  const float* g1wi   = (const float*)d_in[16]; const float* g1wh   = (const float*)d_in[17];
  const float* g1bi   = (const float*)d_in[18]; const float* g1bh   = (const float*)d_in[19];
  const float* g2wi   = (const float*)d_in[20]; const float* g2wh   = (const float*)d_in[21];
  const float* g2bi   = (const float*)d_in[22]; const float* g2bh   = (const float*)d_in[23];
  const float* mel_w  = (const float*)d_in[24]; const float* mel_b  = (const float*)d_in[25];

  char* p = (char*)d_ws;
  auto carve = [&](size_t bytes)->char*{ char* r = p; p += (bytes + 255) & ~((size_t)255); return r; };
  // bf16 weights
  u16* w_pre1 = (u16*)carve((size_t)256*416*2);
  u16* w_pre2 = (u16*)carve((size_t)128*256*2);
  u16* w_awi  = (u16*)carve((size_t)768*384*2);
  u16* w_awh  = (u16*)carve((size_t)768*256*2);
  u16* w_q    = (u16*)carve((size_t)256*256*2);
  u16* w_mem  = (u16*)carve((size_t)256*256*2);
  u16* w_proj = (u16*)carve((size_t)256*512*2);
  u16* w_g1wi = (u16*)carve((size_t)768*256*2);
  u16* w_g1wh = (u16*)carve((size_t)768*256*2);
  u16* w_g2wi = (u16*)carve((size_t)768*256*2);
  u16* w_g2wh = (u16*)carve((size_t)768*256*2);
  u16* w_mel  = (u16*)carve((size_t)400*256*2);
  // big bf16 tensors
  u16* enc_bf   = (u16*)carve((size_t)64*1024*256*2);
  u16* pm_bf    = (u16*)carve((size_t)64*1024*256*2);
  u16* decin_bf = (u16*)carve((size_t)25600*416*2);
  u16* act1_bf  = (u16*)carve((size_t)25600*256*2);
  u16* X_bf     = (u16*)carve((size_t)25600*128*2);
  // recurrent states (contiguous so one zero pass covers them)
  char* stateBase = p;
  float* h_att  = (float*)carve((size_t)64*256*4);
  float* h1     = (float*)carve((size_t)64*256*4);
  float* h2     = (float*)carve((size_t)64*256*4);
  float* attn_f = (float*)carve((size_t)64*256*4);
  u16* h_att_bf = (u16*)carve((size_t)64*256*2);
  u16* h1_bf    = (u16*)carve((size_t)64*256*2);
  u16* h2_bf    = (u16*)carve((size_t)64*256*2);
  u16* attn_bf  = (u16*)carve((size_t)64*256*2);
  int stateWords = (int)(((char*)p - stateBase) / 4);
  // per-step scratch
  float* gi   = (float*)carve((size_t)64*768*4);
  float* gh   = (float*)carve((size_t)64*768*4);
  float* qbuf = (float*)carve((size_t)64*256*4);
  float* decb = (float*)carve((size_t)64*256*4);
  u16*  dec_bf= (u16*)carve((size_t)64*256*2);

  auto conv = [&](const float* s, u16* d, int rows, int K, int Kpad){
    size_t tot = (size_t)rows * Kpad;
    int grid = (int)((tot + 255) / 256); if (grid > 8192) grid = 8192;
    convert_pad_kernel<<<grid, 256, 0, stream>>>(s, d, rows, K, Kpad);
  };

  // ---- precompute (parallel) ----
  conv(pre_w1, w_pre1, 256, 400, 416);
  conv(pre_w2, w_pre2, 128, 256, 256);
  conv(agru_wi, w_awi, 768, 384, 384);
  conv(agru_wh, w_awh, 768, 256, 256);
  conv(attn_qw, w_q,   256, 256, 256);
  conv(mem_w,  w_mem,  256, 256, 256);
  conv(proj_w, w_proj, 256, 512, 512);
  conv(g1wi, w_g1wi, 768, 256, 256);
  conv(g1wh, w_g1wh, 768, 256, 256);
  conv(g2wi, w_g2wi, 768, 256, 256);
  conv(g2wh, w_g2wh, 768, 256, 256);
  conv(mel_w, w_mel, 400, 256, 256);
  conv(enc, enc_bf, 64 * 1024, 256, 256);
  decin_kernel<<<8192, 256, 0, stream>>>(inputs, decin_bf);
  // processed_memory (bf16): (65536x256) @ mem_w.T  -> 65536 tiles
  wmma_gemm_kernel<8, 0><<<(4096 * 16 + 7) / 8, 256, 0, stream>>>(
      enc_bf, 256, nullptr, 0, w_mem, 256, nullptr,
      nullptr, 0, pm_bf, 256, 4096, 16, 0);
  // prenet over all steps
  wmma_gemm_kernel<13, 0><<<(1600 * 16 + 7) / 8, 256, 0, stream>>>(
      decin_bf, 416, nullptr, 0, w_pre1, 416, pre_b1,
      nullptr, 0, act1_bf, 256, 1600, 16, 1);
  wmma_gemm_kernel<8, 0><<<(1600 * 8 + 7) / 8, 256, 0, stream>>>(
      act1_bf, 256, nullptr, 0, w_pre2, 256, pre_b2,
      nullptr, 0, X_bf, 128, 1600, 8, 1);
  // zero recurrent states
  zero_kernel<<<(stateWords + 255) / 256, 256, 0, stream>>>((unsigned*)stateBase, stateWords);

  float* melOut   = (float*)d_out;                          // (B, 400, 400), row b stride 160000
  float* alignOut = (float*)d_out + (size_t)64 * 400 * 400; // (B, 400, 1024)

  // ---- sequential decode ----
  for (int t = 0; t < 400; ++t){
    const u16* Xt = X_bf + (size_t)t * 64 * 128;
    // attention GRU gates: gi = [x_t | attn] @ wi.T + bi ; gh = h_att @ wh.T + bh
    wmma_gates_kernel<4, 8, 8><<<48, 256, 0, stream>>>(
        Xt, 128, attn_bf, 256, w_awi, agru_bi,
        h_att_bf, w_awh, agru_bh, gi, gh, 4, 48);
    gru_update_kernel<<<64, 256, 0, stream>>>(gi, gh, h_att, h_att_bf, nullptr, nullptr);
    // q = h_att @ attn_qw.T
    wmma_gemm_kernel<8, 0><<<(4 * 16 + 7) / 8, 256, 0, stream>>>(
        h_att_bf, 256, nullptr, 0, w_q, 256, nullptr,
        qbuf, 256, nullptr, 0, 4, 16, 0);
    attention_kernel<<<64, 256, 0, stream>>>(qbuf, pm_bf, enc_bf, attn_v, mlen,
                                             alignOut + (size_t)t * 1024, attn_f, attn_bf);
    // dec = [h_att | attn] @ proj_w.T + proj_b
    wmma_gemm_kernel<8, 8><<<(4 * 16 + 7) / 8, 256, 0, stream>>>(
        h_att_bf, 256, attn_bf, 256, w_proj, 512, proj_b,
        decb, 256, dec_bf, 256, 4, 16, 0);
    // GRU1
    wmma_gates_kernel<8, 0, 8><<<48, 256, 0, stream>>>(
        dec_bf, 256, nullptr, 0, w_g1wi, g1bi,
        h1_bf, w_g1wh, g1bh, gi, gh, 4, 48);
    gru_update_kernel<<<64, 256, 0, stream>>>(gi, gh, h1, h1_bf, decb, dec_bf);
    // GRU2
    wmma_gates_kernel<8, 0, 8><<<48, 256, 0, stream>>>(
        dec_bf, 256, nullptr, 0, w_g2wi, g2bi,
        h2_bf, w_g2wh, g2bh, gi, gh, 4, 48);
    gru_update_kernel<<<64, 256, 0, stream>>>(gi, gh, h2, h2_bf, decb, dec_bf);
    // mel: out[b,t,:] = dec @ mel_w.T + mel_b
    wmma_gemm_kernel<8, 0><<<(4 * 25 + 7) / 8, 256, 0, stream>>>(
        dec_bf, 256, nullptr, 0, w_mel, 256, mel_b,
        melOut + (size_t)t * 400, 160000, nullptr, 0, 4, 25, 0);
  }
}
